// AttentionLayer_8589934592381
// MI455X (gfx1250) — matrix operand order repaired and checked by reading, not compiled, not run on hardware
//
#include <hip/hip_runtime.h>
#include <hip/hip_bf16.h>

#define SEQ  2048
#define HID  1024
#define LDSK 1032   // 1024 + 8 bf16 pad to break LDS bank conflicts (row stride 2064B)

typedef __attribute__((ext_vector_type(16))) __bf16 v16bf;
typedef __attribute__((ext_vector_type(8)))  __bf16 v8bf;
typedef __attribute__((ext_vector_type(4)))  __bf16 v4bf;
typedef __attribute__((ext_vector_type(8)))  float  v8f;
typedef __attribute__((ext_vector_type(4)))  unsigned int u32x4;
typedef __attribute__((ext_vector_type(8)))  int  i32x8;
typedef __attribute__((ext_vector_type(4)))  int  i32x4;

#if defined(__has_builtin)
#  if __has_builtin(__builtin_amdgcn_tensor_load_to_lds)
#    define USE_TDM 1
#  endif
#endif
#ifndef USE_TDM
#  define USE_TDM 0
#endif

#define A_BYTES   (32 * LDSK * 2)      // 66048: bf16 A tile
#define SRED_OFF  A_BYTES              // 128B score-reduction buffer
#define F_OFF     (A_BYTES + 128)      // 64KB fp32 TDM staging chunk (16 rows)

// Branch-free tanh: copysign(1 - 2e/(1+e), x), e = exp(-2|x|).
// v_exp_f32 + v_rcp_f32, no EXEC divergence (ocml tanhf branches per lane).
__device__ __forceinline__ float fast_tanh(float x) {
    float ax = __builtin_fabsf(x);
    float e  = __expf(-2.0f * ax);
    float t  = 1.0f - 2.0f * e * __builtin_amdgcn_rcpf(1.0f + e);
    return __builtin_copysignf(t, x);
}

__device__ __forceinline__ void wait_tensorcnt0() {
#if defined(__has_builtin) && __has_builtin(__builtin_amdgcn_s_wait_tensorcnt)
    __builtin_amdgcn_s_wait_tensorcnt(0);
#else
    asm volatile("s_wait_tensorcnt 0x0" ::: "memory");
#endif
}

#if USE_TDM
// TDM: DMA a (rows x 1024) fp32 tile of a 2048x1024 tensor into LDS.
// D# per CDNA5 ISA ch.8: group0={count,lds_addr,global_addr,type=2},
// group1={data_size=4B, tensor_dim0=1024, tensor_dim1=2048, tile_dim0=1024,
//         tile_dim1=rows, tensor_dim0_stride=1024}. Groups 2/3 unused (2-D).
__device__ __forceinline__ void tdm_load_tile(const float* gsrc, void* lds_dst, int rows) {
    unsigned long long ga = (unsigned long long)(size_t)gsrc;
    unsigned int lds_off  = (unsigned int)(size_t)lds_dst;  // flat LDS addr: low 32 bits = LDS byte offset
    u32x4 g0;
    g0[0] = 1u;                                              // count=1 (valid), user mode
    g0[1] = lds_off;                                         // lds_addr
    g0[2] = (unsigned int)(ga & 0xffffffffull);              // global_addr[31:0]
    g0[3] = (unsigned int)((ga >> 32) & 0x1ffffffull) | (2u << 30); // global_addr[56:32] | type=2
    i32x8 g1;
    g1[0] = (int)(2u << 16);          // workgroup_mask=0, data_size=2 (4 bytes)
    g1[1] = (int)(1024u << 16);       // [63:48] tensor_dim0[15:0] = 1024
    g1[2] = (int)(2048u << 16);       // [79:64] dim0 hi = 0; [95:80] tensor_dim1[15:0] = 2048
    g1[3] = (int)(1024u << 16);       // [111:96] dim1 hi = 0; [127:112] tile_dim0 = 1024
    g1[4] = rows;                     // [143:128] tile_dim1 = rows; tile_dim2 = 0
    g1[5] = 1024;                     // tensor_dim0_stride[31:0] = 1024 elements
    g1[6] = 0;                        // stride hi, tensor_dim1_stride lo
    g1[7] = 0;
    i32x4 z4 = {0, 0, 0, 0};
#if __clang_major__ >= 23
    i32x8 z8 = {0, 0, 0, 0, 0, 0, 0, 0};
    __builtin_amdgcn_tensor_load_to_lds(g0, g1, z4, z4, z8, 0);
#else
    __builtin_amdgcn_tensor_load_to_lds(g0, g1, z4, z4, 0);
#endif
}
#endif

// --- Kernel A: W0a (H x H, input-major) -> bf16 transposed [n][k] -------------
__global__ void prep_wt_kernel(const float* __restrict__ w0, __bf16* __restrict__ wt) {
    int idx = blockIdx.x * 256 + threadIdx.x;
    int n = idx & (HID - 1);
    int k = idx >> 10;
    wt[(size_t)n * HID + k] = (__bf16)w0[(size_t)k * HID + n];
}

// --- Kernel B: u[b][h] = b0[h] + hidden[b] @ W0b ------------------------------
__global__ void prep_u_kernel(const float* __restrict__ hidden,
                              const float* __restrict__ w0,
                              const float* __restrict__ b0,
                              float* __restrict__ u) {
    int b = blockIdx.x >> 2;
    int h = ((blockIdx.x & 3) << 8) + threadIdx.x;
    const float* hb = hidden + (size_t)b * HID;
    float acc = b0[h];
    for (int k = 0; k < HID; ++k)
        acc = fmaf(hb[k], w0[(size_t)(HID + k) * HID + h], acc);
    u[(size_t)b * HID + h] = acc;
}

// --- Kernel C: fused GEMM(bf16 WMMA) + tanh + dot(w1) -> scores ---------------
__global__ void attn_scores_kernel(const float* __restrict__ enc,
                                   const __bf16* __restrict__ wt,
                                   const float* __restrict__ u,
                                   const float* __restrict__ w1,
                                   const float* __restrict__ b1,
                                   const unsigned char* __restrict__ mask,
                                   float* __restrict__ scores) {
    extern __shared__ char smem_raw[];
    __bf16* A    = (__bf16*)smem_raw;
    float*  sred = (float*)(smem_raw + SRED_OFF);

    const int tid    = threadIdx.x;
    const int lane   = tid & 31;
    const int wave   = tid >> 5;
    const int wave_m = wave >> 2;
    const int wave_n = wave & 3;
    const int b      = blockIdx.x >> 6;
    const int s_base = (blockIdx.x & 63) * 32;

    if (tid < 32) sred[tid] = 0.0f;

    const float* encb = enc + ((size_t)b * SEQ + s_base) * HID;

#if USE_TDM
    // TDM DMA: 2 chunks of 16 rows x 1024 fp32 -> LDS, then convert to bf16 tile.
    float* F = (float*)(smem_raw + F_OFF);
    for (int chunk = 0; chunk < 2; ++chunk) {
        if (wave == 0) {
            tdm_load_tile(encb + (size_t)chunk * 16 * HID, F, 16);
            wait_tensorcnt0();
        }
        __syncthreads();
        for (int i = 0; i < 16; ++i) {
            int e4 = tid + i * 256;              // 4096 float4 per chunk
            int r  = e4 >> 8;
            int k4 = (e4 & 255) << 2;
            float4 v = *(const float4*)(F + r * HID + k4);   // ds_load_b128
            v4bf p;
            p[0] = (__bf16)v.x; p[1] = (__bf16)v.y; p[2] = (__bf16)v.z; p[3] = (__bf16)v.w;
            *(v4bf*)(A + (chunk * 16 + r) * LDSK + k4) = p;
        }
        __syncthreads();
    }
#else
    // Fallback: direct global fp32 -> bf16 LDS staging.
    for (int i = 0; i < 32; ++i) {
        int e4 = tid + i * 256;
        int r  = e4 >> 8;
        int k4 = (e4 & 255) << 2;
        float4 v = *(const float4*)(encb + (size_t)r * HID + k4);
        v4bf p;
        p[0] = (__bf16)v.x; p[1] = (__bf16)v.y; p[2] = (__bf16)v.z; p[3] = (__bf16)v.w;
        *(v4bf*)(A + r * LDSK + k4) = p;
    }
    __syncthreads();
#endif

    const int mloc   = lane & 15;
    const int khalfA = (lane >> 4) << 4;
    const int khalfB = (lane >> 4) << 4;
    const int arow   = wave_m * 16 + mloc;

    float s_acc[8];
#pragma unroll
    for (int r = 0; r < 8; ++r) s_acc[r] = 0.0f;

    for (int ni = 0; ni < 16; ++ni) {
        const int n   = ni * 64 + wave_n * 16 + mloc;
        const float w1v = w1[n];
        const float uv  = u[(size_t)b * HID + n];
        const __bf16* wrow = wt + (size_t)n * HID + khalfB;
        v8f acc = {};
#pragma unroll 4
        for (int kk = 0; kk < 32; ++kk) {
            const int k0 = kk * 32;
            v8bf lo = *(const v8bf*)(A + arow * LDSK + k0 + khalfA);
            v8bf hi = *(const v8bf*)(A + arow * LDSK + k0 + khalfA + 8);
            v16bf af;
#pragma unroll
            for (int i = 0; i < 8; ++i) { af[i] = lo[i]; af[i + 8] = hi[i]; }
            v16bf bf_ = *(const v16bf*)(wrow + k0);
            acc = __builtin_amdgcn_wmma_f32_16x16x32_bf16(
                false, af, false, bf_, (short)0, acc, false, false);
        }
#pragma unroll
        for (int r = 0; r < 8; ++r)
            s_acc[r] += fast_tanh(acc[r] + uv) * w1v;
    }

#pragma unroll
    for (int r = 0; r < 8; ++r) {
        float v = s_acc[r];
        v += __shfl_xor(v, 1);
        v += __shfl_xor(v, 2);
        v += __shfl_xor(v, 4);
        v += __shfl_xor(v, 8);
        if (mloc == 0) {
            int row = wave_m * 16 + r + ((lane >> 4) << 3);
            atomicAdd(&sred[row], v);   // ds_add_f32
        }
    }
    __syncthreads();

    if (tid < 32) {
        int s = s_base + tid;
        float sc = sred[tid] + b1[0];
        if (!mask[(size_t)b * SEQ + s]) sc = -__builtin_inff();
        scores[(size_t)b * SEQ + s] = sc;
    }
}

// --- Kernel D: softmax over S per batch; also zero the (poisoned) output ------
__global__ void softmax_kernel(float* __restrict__ scores, float* __restrict__ out) {
    __shared__ float red[256];
    const int b = blockIdx.x, tid = threadIdx.x;
    float* sc = scores + (size_t)b * SEQ;
    float vals[8];
    float mx = -__builtin_inff();
#pragma unroll
    for (int i = 0; i < 8; ++i) { vals[i] = sc[tid + i * 256]; mx = fmaxf(mx, vals[i]); }
    red[tid] = mx; __syncthreads();
    for (int off = 128; off > 0; off >>= 1) {
        if (tid < off) red[tid] = fmaxf(red[tid], red[tid + off]);
        __syncthreads();
    }
    mx = red[0]; __syncthreads();
    float sum = 0.0f;
#pragma unroll
    for (int i = 0; i < 8; ++i) { vals[i] = __expf(vals[i] - mx); sum += vals[i]; }
    red[tid] = sum; __syncthreads();
    for (int off = 128; off > 0; off >>= 1) {
        if (tid < off) red[tid] += red[tid + off];
        __syncthreads();
    }
    const float inv = 1.0f / red[0];
#pragma unroll
    for (int i = 0; i < 8; ++i) sc[tid + i * 256] = vals[i] * inv;
#pragma unroll
    for (int i = 0; i < 4; ++i) out[(size_t)b * HID + tid + i * 256] = 0.0f;
}

// --- Kernel E: new_hidden = attn @ enc (S-chunked, atomic accumulate) ---------
__global__ void weighted_sum_kernel(const float* __restrict__ enc,
                                    const float* __restrict__ attn,
                                    float* __restrict__ out) {
    const int blk = blockIdx.x;
    const int b  = blk >> 5;
    const int sc = (blk >> 2) & 7;
    const int hc = blk & 3;
    const int h  = (hc << 8) + threadIdx.x;
    const float* encb = enc + (size_t)b * SEQ * HID;
    const float* ab   = attn + (size_t)b * SEQ;
    float acc = 0.0f;
    const int s0 = sc * 256;
    for (int s = s0; s < s0 + 256; ++s)
        acc = fmaf(ab[s], encb[(size_t)s * HID + h], acc);
    atomicAdd(&out[(size_t)b * HID + h], acc);
}

extern "C" void kernel_launch(void* const* d_in, const int* in_sizes, int n_in,
                              void* d_out, int out_size, void* d_ws, size_t ws_size,
                              hipStream_t stream) {
    const float*         hidden = (const float*)d_in[0];
    const float*         enc    = (const float*)d_in[1];
    const unsigned char* mask   = (const unsigned char*)d_in[2];
    const float*         w0     = (const float*)d_in[3];
    const float*         b0     = (const float*)d_in[4];
    const float*         w1     = (const float*)d_in[5];
    const float*         b1     = (const float*)d_in[6];
    float* out = (float*)d_out;

    char* ws = (char*)d_ws;
    __bf16* wt     = (__bf16*)ws;                                         // 2 MB
    float*  u      = (float*)(ws + (size_t)HID * HID * 2);                // 128 KB
    float*  scores = (float*)(ws + (size_t)HID * HID * 2 + 32 * HID * 4); // 256 KB

    prep_wt_kernel<<<(HID * HID) / 256, 256, 0, stream>>>(w0, wt);
    prep_u_kernel<<<128, 256, 0, stream>>>(hidden, w0, b0, u);

#if USE_TDM
    const size_t shmem = F_OFF + 16 * HID * sizeof(float);   // ~128.6 KB (2 WG / 320KB WGP)
#else
    const size_t shmem = A_BYTES + 32 * sizeof(float);
#endif
    attn_scores_kernel<<<32 * 64, 256, shmem, stream>>>(enc, wt, u, w1, b1, mask, scores);

    softmax_kernel<<<32, 256, 0, stream>>>(scores, out);
    weighted_sum_kernel<<<32 * 8 * 4, 256, 0, stream>>>(enc, scores, out);
}
